// GPN_22162031247581
// MI455X (gfx1250) — matrix operand order repaired and checked by reading, not compiled, not run on hardware
//
#include <hip/hip_runtime.h>
#include <hip/hip_bf16.h>
#include <math.h>
#include <stdint.h>

// ---------------------------------------------------------------------------
// CDNA5 (gfx1250) implementation. wave32. All GEMMs via
// v_wmma_f32_16x16x32_bf16 (bf16 inputs, f32 accumulate). K is always 256.
// Weights pre-transposed to bf16 Wt[N][K] -> two 16B B-frag loads
// per lane. Activations are staged as bf16 in LDS (converted once), so
// A-fragments are two ds_load_b128 with no per-use conversion. The LSTM
// streams bf16 xs tiles with the Tensor Data Mover, double-buffered,
// synced via s_wait_tensorcnt. kc loops are kept rolled to avoid the
// hoisted-B-load register spills seen in earlier rounds.
// ---------------------------------------------------------------------------

typedef __bf16 bf16;
typedef __attribute__((ext_vector_type(16))) __bf16 v16bf;
typedef __attribute__((ext_vector_type(8)))  float  v8f;
typedef __attribute__((ext_vector_type(4)))  unsigned int v4u;
typedef __attribute__((ext_vector_type(8)))  int v8i;
typedef __attribute__((ext_vector_type(4)))  int v4i;

#define DEV __device__ __forceinline__

constexpr int kDIM = 256;
constexpr int kB   = 256;
constexpr int kS   = 512;

DEV float sigm(float x) { return 1.f / (1.f + expf(-x)); }

DEV v8f wmma_bf16(v16bf a, v16bf b, v8f c) {
  // (neg_a, A, neg_b, B, c_mod, C, reuse_a, reuse_b)
  return __builtin_amdgcn_wmma_f32_16x16x32_bf16(false, a, false, b,
                                                 (short)0, c, false, false);
}

DEV unsigned pack2bf(float a, float b) {
  union { bf16 h[2]; unsigned u; } p;
  p.h[0] = (bf16)a;
  p.h[1] = (bf16)b;
  return p.u;
}

// A-matrix 16x32 bf16 fragment from an f32 row (with conversion).
// ISA layout: lane half h: element e -> K = kbase + e + 8*h + (e>=8 ? 8 : 0).
DEV v16bf afrag_f32(const float* rowp, int kbase, int half) {
  const float* p0 = rowp + kbase + 8 * half;
  const float* p1 = rowp + kbase + 16 + 8 * half;
  v16bf a;
#pragma unroll
  for (int i = 0; i < 8; ++i) a[i] = (bf16)p0[i];
#pragma unroll
  for (int i = 0; i < 8; ++i) a[8 + i] = (bf16)p1[i];
  return a;
}

// A-matrix fragment from a bf16 row (LDS or global): two 16B loads, no cvt.
DEV v16bf afrag_bf16(const bf16* rowp, int kbase, int half) {
  const bf16* p0 = rowp + kbase + 8 * half;
  const bf16* p1 = rowp + kbase + 16 + 8 * half;
  v16bf a;
#pragma unroll
  for (int i = 0; i < 8; ++i) a[i] = p0[i];
#pragma unroll
  for (int i = 0; i < 8; ++i) a[8 + i] = p1[i];
  return a;
}

// B-matrix 32x16 bf16 fragment from transposed weights Wt[N][256].
// B striping matches A: lane half h: element e -> K = kc*32 + e + 8*h + (e>=8 ? 8 : 0),
// column n = lane&15 -> two 16B loads per lane.
DEV v16bf bfrag(const bf16* Wt, int n, int kc, int half) {
  const bf16* p = Wt + (size_t)n * kDIM + kc * 32 + 8 * half;
  v16bf b;
#pragma unroll
  for (int i = 0; i < 8; ++i) b[i] = p[i];
#pragma unroll
  for (int i = 0; i < 8; ++i) b[8 + i] = p[16 + i];
  return b;
}

// ---------------------------------------------------------------------------
// TDM: load a 2D tile (tile_w x tile_h elements of elem_code size, row stride
// `stride_elems`) from global into LDS at byte offset lds_off.
// D# bit packing per CDNA5 ISA ch. 8 (group0 128b, group1 256b).
// elem_code: 0=1B 1=2B 2=4B 3=8B.
// ---------------------------------------------------------------------------
DEV void tdm_load_tile(const void* gsrc, unsigned lds_off, unsigned elem_code,
                       unsigned tile_w, unsigned tile_h,
                       unsigned stride_elems) {
  unsigned long long ga = (unsigned long long)(uintptr_t)gsrc;
  v4u g0;
  g0[0] = 1u;                                     // count=1, user mode
  g0[1] = lds_off;                                // lds_addr (bytes)
  g0[2] = (unsigned)(ga & 0xFFFFFFFFu);           // global_addr[31:0]
  g0[3] = (unsigned)((ga >> 32) & 0x1FFFFFFu)     // global_addr[56:32]
          | (2u << 30);                           // type=2 ("image")
  v8i g1;
  g1[0] = (int)(elem_code << 16);                 // data_size
  g1[1] = (int)((tile_w & 0xFFFFu) << 16);        // tensor_dim0[15:0]
  g1[2] = (int)((tile_w >> 16) | ((tile_h & 0xFFFFu) << 16));  // d0 hi|d1 lo
  g1[3] = (int)((tile_h >> 16) | (tile_w << 16)); // d1 hi | tile_dim0
  g1[4] = (int)(tile_h & 0xFFFFu);                // tile_dim1 | tile_dim2=0
  g1[5] = (int)stride_elems;                      // tensor_dim0_stride lo32
  g1[6] = 0;                                      // stride0 hi16 | stride1 lo16
  g1[7] = 0;
  v4i z4 = {0, 0, 0, 0};
#if defined(__clang_major__) && (__clang_major__ >= 23)
  v8i z8 = {0, 0, 0, 0, 0, 0, 0, 0};
  __builtin_amdgcn_tensor_load_to_lds(g0, g1, z4, z4, z8, 0);
#else
  __builtin_amdgcn_tensor_load_to_lds(g0, g1, z4, z4, 0);
#endif
}

// ---------------------------------------------------------------------------
// Weight prep: f32 W[K][N] -> bf16 Wt[N][K]
// ---------------------------------------------------------------------------
__global__ __launch_bounds__(256) void k_wt(const float* __restrict__ W,
                                            bf16* __restrict__ Wt,
                                            int K, int N) {
  int idx = blockIdx.x * 256 + threadIdx.x;
  if (idx >= K * N) return;
  int n = idx / K, k = idx % K;
  Wt[idx] = (bf16)W[(size_t)k * N + n];
}

// ---------------------------------------------------------------------------
// Embedding (FEAT=2): out[row][d] = e0*W[0][d] + e1*W[1][d] + b[d]  (* nz)
// Optionally also writes a bf16 copy (LSTM GEMM operand).
// ---------------------------------------------------------------------------
__global__ __launch_bounds__(256) void k_embed(const float* __restrict__ X2,
                                               const float* __restrict__ W,
                                               const float* __restrict__ bias,
                                               float* __restrict__ out,
                                               bf16* __restrict__ outbf,
                                               float* __restrict__ nzArr,
                                               int applyMask) {
  int idx = blockIdx.x * 256 + threadIdx.x;
  int row = idx >> 8, d = idx & 255;
  float e0 = X2[row * 2 + 0], e1 = X2[row * 2 + 1];
  float val = e0 * W[d] + e1 * W[kDIM + d] + bias[d];
  if (applyMask) {
    float nzv = ((e0 + e1) != 0.f) ? 1.f : 0.f;
    val *= nzv;
    if (d == 0) nzArr[row] = nzv;
  }
  out[idx] = val;
  if (outbf) outbf[idx] = (bf16)val;
}

__global__ __launch_bounds__(256) void k_ntrue(const float* __restrict__ nz,
                                               float* __restrict__ ntr) {
  __shared__ float red[256];
  int b = blockIdx.x, tid = threadIdx.x;
  float s = 0.f;
  for (int i = tid; i < kS; i += 256) s += nz[(size_t)b * kS + i];
  red[tid] = s;
  __syncthreads();
  for (int w = 128; w > 0; w >>= 1) {
    if (tid < w) red[tid] += red[tid + w];
    __syncthreads();
  }
  if (tid == 0) ntr[b] = red[0];
}

// ---------------------------------------------------------------------------
// LSTM scan. 16 blocks x 16 batch rows, h/c resident in LDS, 512 local steps.
// Per-step bf16 xs tile (16 x 256) TDM-prefetched into LDS, double-buffered.
// h has a bf16 shadow in LDS, so GEMM A-fragments are pure ds_load_b128.
// ---------------------------------------------------------------------------
__global__ __launch_bounds__(256) void k_lstm(
    const bf16* __restrict__ ctxbf, const float* __restrict__ nz,
    const bf16* __restrict__ Wih_t, const bf16* __restrict__ Whh_t,
    const float* __restrict__ b_ih, const float* __restrict__ b_hh,
    const float* __restrict__ h0, const float* __restrict__ c0,
    float* __restrict__ h_out, float* __restrict__ c_out) {
  __shared__ bf16 sxs[2][16 * 256];   // TDM-staged xs tiles (8KB each)
  __shared__ bf16 shb[16 * 256];      // bf16 shadow of h (8KB)
  __shared__ float sh[16 * 256];      // h state (16KB)
  __shared__ float sc[16 * 256];      // c state (16KB)
  __shared__ float bufA[16 * 256];    // gate scratch (16KB)
  __shared__ float bufB[16 * 256];    // gate scratch (16KB)
  const int tid = threadIdx.x;
  const int wave = tid >> 5, lane = tid & 31;
  const int m = lane & 15, half = lane >> 4;
  const int b0 = blockIdx.x * 16;

  for (int i = tid; i < 16 * 256; i += 256) {
    float hv = h0[i & 255];
    sh[i] = hv;
    shb[i] = (bf16)hv;
    sc[i] = c0[i & 255];
  }
  const unsigned lds_xs0 = (unsigned)(uintptr_t)(&sxs[0][0]);
  const unsigned lds_xs1 = (unsigned)(uintptr_t)(&sxs[1][0]);
  // prefetch tile for t=0
  if (wave == 0) {
    tdm_load_tile(ctxbf + (size_t)b0 * kS * kDIM, lds_xs0, 1u, kDIM, 16,
                  (unsigned)(kS * kDIM));
  }
  __syncthreads();

  for (int t = 0; t < kS; ++t) {
    if (wave == 0) __builtin_amdgcn_s_wait_tensorcnt(0);
    __syncthreads();  // tile[t&1] visible to all waves
    if (wave == 0 && (t + 1) < kS) {
      tdm_load_tile(ctxbf + ((size_t)b0 * kS + (t + 1)) * kDIM,
                    ((t + 1) & 1) ? lds_xs1 : lds_xs0, 1u, kDIM, 16,
                    (unsigned)(kS * kDIM));
    }
    const bf16* xrow = &sxs[t & 1][m * 256];
    const bf16* hrow = &shb[m * 256];

    // gate = 0:i 1:f 2:g 3:o; each wave owns cols [wave*32, wave*32+32)
    auto gate_gemm = [&](int gate, float* buf) {
#pragma unroll 1
      for (int ct = 0; ct < 2; ++ct) {
        int ncol = wave * 32 + ct * 16 + m;  // col within gate (0..255)
        int n = gate * 256 + ncol;           // col within 1024
        float bv = b_ih[n] + b_hh[n];
        v8f acc;
#pragma unroll
        for (int r = 0; r < 8; ++r) acc[r] = bv;
#pragma unroll 1
        for (int kc = 0; kc < 8; ++kc) {
          acc = wmma_bf16(afrag_bf16(xrow, kc * 32, half),
                          bfrag(Wih_t, n, kc, half), acc);
          acc = wmma_bf16(afrag_bf16(hrow, kc * 32, half),
                          bfrag(Whh_t, n, kc, half), acc);
        }
#pragma unroll
        for (int r = 0; r < 8; ++r)
          buf[(r + 8 * half) * 256 + ncol] = acc[r];
      }
    };

    gate_gemm(0, bufA);  // raw i
    gate_gemm(2, bufB);  // raw g
    __syncthreads();
    for (int i = tid; i < 4096; i += 256)
      bufA[i] = sigm(bufA[i]) * tanhf(bufB[i]);  // t1 = sig(i)*tanh(g)
    __syncthreads();
    gate_gemm(1, bufB);  // raw f
    __syncthreads();
    {
      int row = tid >> 4, col0 = (tid & 15) * 16;
      float nzr = nz[(size_t)(b0 + row) * kS + t];
      if (nzr != 0.f) {
#pragma unroll
        for (int j = 0; j < 16; ++j) {
          int idx = row * 256 + col0 + j;
          sc[idx] = sigm(bufB[idx]) * sc[idx] + bufA[idx];
        }
      }
    }
    __syncthreads();
    gate_gemm(3, bufB);  // raw o (shb not yet updated -> old h, correct)
    __syncthreads();
    {
      int row = tid >> 4, col0 = (tid & 15) * 16;
      float nzr = nz[(size_t)(b0 + row) * kS + t];
      if (nzr != 0.f) {
#pragma unroll
        for (int j = 0; j < 16; ++j) {
          int idx = row * 256 + col0 + j;
          float hv = sigm(bufB[idx]) * tanhf(sc[idx]);
          sh[idx] = hv;
          shb[idx] = (bf16)hv;
        }
      }
    }
    __syncthreads();
  }

  for (int i = tid; i < 4096; i += 256) {
    size_t g = (size_t)(b0 + (i >> 8)) * kDIM + (i & 255);
    h_out[g] = sh[i];
    c_out[g] = sc[i];
  }
}

// ---------------------------------------------------------------------------
// Stage 128 rows (f32, optional per-row scale) into a 64KB bf16 LDS tile.
// Coalesced float4 reads, packed bf16 (uint2) LDS writes.
// ---------------------------------------------------------------------------
DEV void stage_rows_bf16(bf16* sA, const float* __restrict__ src,
                         int rowBase, const float* __restrict__ rowScale,
                         int tid) {
  for (int e4 = tid; e4 < 128 * 64; e4 += 256) {  // 8192 float4s
    int row = e4 >> 6;  // 64 float4 per row
    float s = rowScale ? rowScale[rowBase + row] : 1.f;
    float4 v = ((const float4*)src)[(size_t)rowBase * 64 + e4];
    uint2 pk;
    pk.x = pack2bf(v.x * s, v.y * s);
    pk.y = pack2bf(v.z * s, v.w * s);
    ((uint2*)sA)[e4] = pk;
  }
}

// ---------------------------------------------------------------------------
// Agg round, IN-PLACE on ctx:
//   cm = ctx*nz ; ctx = (r*(cm@W+bW) + (1-r)*relu((cm@A)/den + bA)) * nz
// Input rows staged to LDS (bf16) before any store -> in-place is safe.
// ---------------------------------------------------------------------------
__global__ __launch_bounds__(256) void k_agg(
    float* ctx, const bf16* __restrict__ Wt, const float* __restrict__ bW,
    const bf16* __restrict__ At, const float* __restrict__ bA,
    const float* __restrict__ rp, const float* __restrict__ nz,
    const float* __restrict__ ntr) {
  __shared__ bf16 sA[128 * 256];  // 64KB staged cm rows
  const int tid = threadIdx.x;
  const int wave = tid >> 5, lane = tid & 31;
  const int m = lane & 15, half = lane >> 4;
  const int blockRow = blockIdx.x * 128;
  const int rowBase = blockRow + wave * 16;

  stage_rows_bf16(sA, ctx, blockRow, nz, tid);
  __syncthreads();

  const bf16* arow = &sA[(wave * 16 + m) * 256];
  float r = rp[0], omr = 1.f - r;
  float nzr[8], dinv[8];
#pragma unroll
  for (int rr = 0; rr < 8; ++rr) {
    int rowr = rowBase + rr + 8 * half;
    nzr[rr] = nz[rowr];
    dinv[rr] = 1.f / (ntr[rowr / kS] - 1.f);
  }

#pragma unroll 1
  for (int ct = 0; ct < 16; ++ct) {
    int n = ct * 16 + m;
    v8f P, Q;
    float bWn = bW[n], bAn = bA[n];
#pragma unroll
    for (int i = 0; i < 8; ++i) { P[i] = bWn; Q[i] = 0.f; }
#pragma unroll 1
    for (int kc = 0; kc < 8; ++kc) {
      v16bf a = afrag_bf16(arow, kc * 32, half);
      P = wmma_bf16(a, bfrag(Wt, n, kc, half), P);
      Q = wmma_bf16(a, bfrag(At, n, kc, half), Q);
    }
#pragma unroll
    for (int rr = 0; rr < 8; ++rr) {
      float q = Q[rr] * dinv[rr] + bAn;
      float outv = (r * P[rr] + omr * fmaxf(q, 0.f)) * nzr[rr];
      ctx[(size_t)(rowBase + rr + 8 * half) * kDIM + n] = outv;
    }
  }
}

// ---------------------------------------------------------------------------
// Generic R x 256 @ 256 x N GEMM (+bias), rows staged to LDS. Used for qp.
// ---------------------------------------------------------------------------
__global__ __launch_bounds__(256) void k_gemm(const float* __restrict__ IN,
                                              const bf16* __restrict__ Wt,
                                              const float* __restrict__ bias,
                                              float* __restrict__ OUT, int N) {
  __shared__ bf16 sA[128 * 256];
  const int tid = threadIdx.x;
  const int wave = tid >> 5, lane = tid & 31;
  const int m = lane & 15, half = lane >> 4;
  const int blockRow = blockIdx.x * 128;
  const int rowBase = blockRow + wave * 16;

  stage_rows_bf16(sA, IN, blockRow, nullptr, tid);
  __syncthreads();

  const bf16* arow = &sA[(wave * 16 + m) * 256];
#pragma unroll 1
  for (int ct = 0; ct * 16 < N; ++ct) {
    int n = ct * 16 + m;
    float bv = bias ? bias[n] : 0.f;
    v8f acc;
#pragma unroll
    for (int r = 0; r < 8; ++r) acc[r] = bv;
#pragma unroll 1
    for (int kc = 0; kc < 8; ++kc)
      acc = wmma_bf16(afrag_bf16(arow, kc * 32, half),
                      bfrag(Wt, n, kc, half), acc);
#pragma unroll
    for (int r = 0; r < 8; ++r)
      OUT[(size_t)(rowBase + r + 8 * half) * N + n] = acc[r];
  }
}

// ---------------------------------------------------------------------------
// Encoder part 1: c_new = sig(F)*c + sig(I)*tanh(G); F/I: 3 GEMMs, G: 2.
// Rolled kc loop, fragments rebuilt from L1-hot rows -> low VGPR pressure.
// ---------------------------------------------------------------------------
__global__ __launch_bounds__(256) void k_enc1(
    const float* __restrict__ xe, const float* __restrict__ hl,
    const float* __restrict__ cl,
    const bf16* Wxi, const float* bxi, const bf16* Whi, const float* bhi,
    const bf16* Wci, const float* bci,
    const bf16* Wxf, const float* bxf, const bf16* Whf, const float* bhf,
    const bf16* Wcf, const float* bcf,
    const bf16* Wxc, const float* bxc, const bf16* Whc, const float* bhc,
    float* __restrict__ c_new) {
  const int tid = threadIdx.x;
  const int wave = tid >> 5, lane = tid & 31;
  const int m = lane & 15, half = lane >> 4;
  const int rowBase = blockIdx.x * 128 + wave * 16;
  const float* px = xe + (size_t)(rowBase + m) * kDIM;
  const float* ph = hl + (size_t)(rowBase + m) * kDIM;
  const float* pc = cl + (size_t)(rowBase + m) * kDIM;
#pragma unroll 1
  for (int ct = 0; ct < 16; ++ct) {
    int n = ct * 16 + m;
    float bi = bxi[n] + bhi[n] + bci[n];
    float bff = bxf[n] + bhf[n] + bcf[n];
    float bg = bxc[n] + bhc[n];
    v8f ai, af, ag;
#pragma unroll
    for (int r = 0; r < 8; ++r) { ai[r] = bi; af[r] = bff; ag[r] = bg; }
#pragma unroll 1
    for (int kc = 0; kc < 8; ++kc) {
      v16bf ax = afrag_f32(px, kc * 32, half);
      v16bf ah = afrag_f32(ph, kc * 32, half);
      v16bf ac = afrag_f32(pc, kc * 32, half);
      ai = wmma_bf16(ax, bfrag(Wxi, n, kc, half), ai);
      ai = wmma_bf16(ah, bfrag(Whi, n, kc, half), ai);
      ai = wmma_bf16(ac, bfrag(Wci, n, kc, half), ai);
      af = wmma_bf16(ax, bfrag(Wxf, n, kc, half), af);
      af = wmma_bf16(ah, bfrag(Whf, n, kc, half), af);
      af = wmma_bf16(ac, bfrag(Wcf, n, kc, half), af);
      ag = wmma_bf16(ax, bfrag(Wxc, n, kc, half), ag);
      ag = wmma_bf16(ah, bfrag(Whc, n, kc, half), ag);
    }
#pragma unroll
    for (int r = 0; r < 8; ++r) {
      size_t rr = rowBase + r + 8 * half;
      float co = cl[rr * kDIM + n];
      c_new[rr * kDIM + n] = sigm(af[r]) * co + sigm(ai[r]) * tanhf(ag[r]);
    }
  }
}

// Encoder part 2: h_new = sig(O)*tanh(c_new), O = 3 GEMMs (x, h, c_new).
__global__ __launch_bounds__(256) void k_enc2(
    const float* __restrict__ xe, const float* __restrict__ hl,
    const float* __restrict__ cn,
    const bf16* Wxo, const float* bxo, const bf16* Who, const float* bho,
    const bf16* Wco, const float* bco, float* __restrict__ h_new) {
  const int tid = threadIdx.x;
  const int wave = tid >> 5, lane = tid & 31;
  const int m = lane & 15, half = lane >> 4;
  const int rowBase = blockIdx.x * 128 + wave * 16;
  const float* px = xe + (size_t)(rowBase + m) * kDIM;
  const float* ph = hl + (size_t)(rowBase + m) * kDIM;
  const float* pc = cn + (size_t)(rowBase + m) * kDIM;
#pragma unroll 1
  for (int ct = 0; ct < 16; ++ct) {
    int n = ct * 16 + m;
    float bo = bxo[n] + bho[n] + bco[n];
    v8f ao;
#pragma unroll
    for (int r = 0; r < 8; ++r) ao[r] = bo;
#pragma unroll 1
    for (int kc = 0; kc < 8; ++kc) {
      ao = wmma_bf16(afrag_f32(px, kc * 32, half),
                     bfrag(Wxo, n, kc, half), ao);
      ao = wmma_bf16(afrag_f32(ph, kc * 32, half),
                     bfrag(Who, n, kc, half), ao);
      ao = wmma_bf16(afrag_f32(pc, kc * 32, half),
                     bfrag(Wco, n, kc, half), ao);
    }
#pragma unroll
    for (int r = 0; r < 8; ++r) {
      size_t rr = rowBase + r + 8 * half;
      h_new[rr * kDIM + n] = sigm(ao[r]) * tanhf(cn[rr * kDIM + n]);
    }
  }
}

// ---------------------------------------------------------------------------
// Fused refp + attention score:
//   latent[row] = sum_d v[d] * tanh(qp[b][d] + (ctxf_row @ Wref + bref)[d])
// refp never hits memory. Cross-lane sum via shfl_xor within each 16-lane half.
// ---------------------------------------------------------------------------
__global__ __launch_bounds__(256) void k_uref(const float* __restrict__ ctxf,
                                              const bf16* __restrict__ Wreft,
                                              const float* __restrict__ bref,
                                              const float* __restrict__ qp,
                                              const float* __restrict__ v,
                                              float* __restrict__ latent) {
  __shared__ bf16 sA[128 * 256];
  const int tid = threadIdx.x;
  const int wave = tid >> 5, lane = tid & 31;
  const int m = lane & 15, half = lane >> 4;
  const int blockRow = blockIdx.x * 128;
  const int rowBase = blockRow + wave * 16;

  stage_rows_bf16(sA, ctxf, blockRow, nullptr, tid);
  __syncthreads();

  const bf16* arow = &sA[(wave * 16 + m) * 256];
  float part[8];
#pragma unroll
  for (int r = 0; r < 8; ++r) part[r] = 0.f;

#pragma unroll 1
  for (int ct = 0; ct < 16; ++ct) {
    int n = ct * 16 + m;
    float bv = bref[n], vn = v[n];
    v8f acc;
#pragma unroll
    for (int r = 0; r < 8; ++r) acc[r] = bv;
#pragma unroll 1
    for (int kc = 0; kc < 8; ++kc)
      acc = wmma_bf16(afrag_bf16(arow, kc * 32, half),
                      bfrag(Wreft, n, kc, half), acc);
#pragma unroll
    for (int r = 0; r < 8; ++r) {
      int rr = rowBase + r + 8 * half;
      int b = rr / kS;
      part[r] += vn * tanhf(qp[(size_t)b * kDIM + n] + acc[r]);
    }
  }
#pragma unroll
  for (int r = 0; r < 8; ++r) {
    for (int off = 1; off < 16; off <<= 1)
      part[r] += __shfl_xor(part[r], off, 32);
  }
  if (m == 0) {
#pragma unroll
    for (int r = 0; r < 8; ++r) latent[rowBase + r + 8 * half] = part[r];
  }
}

// Softmax over S=512: u = 10*tanh(latent) + mask
__global__ __launch_bounds__(256) void k_softmax(const float* __restrict__ lat,
                                                 const float* __restrict__ mask,
                                                 float* __restrict__ out) {
  __shared__ float red[256];
  int b = blockIdx.x, tid = threadIdx.x;
  size_t base = (size_t)b * kS;
  float u0 = 10.f * tanhf(lat[base + tid]) + mask[base + tid];
  float u1 = 10.f * tanhf(lat[base + tid + 256]) + mask[base + tid + 256];
  red[tid] = fmaxf(u0, u1);
  __syncthreads();
  for (int w = 128; w > 0; w >>= 1) {
    if (tid < w) red[tid] = fmaxf(red[tid], red[tid + w]);
    __syncthreads();
  }
  float mx = red[0];
  __syncthreads();
  float e0 = expf(u0 - mx), e1 = expf(u1 - mx);
  red[tid] = e0 + e1;
  __syncthreads();
  for (int w = 128; w > 0; w >>= 1) {
    if (tid < w) red[tid] += red[tid + w];
    __syncthreads();
  }
  float inv = 1.f / red[0];
  out[base + tid] = e0 * inv;
  out[base + tid + 256] = e1 * inv;
}

// ---------------------------------------------------------------------------
// Input indices per setup_inputs() insertion order (lin -> (W, b))
// ---------------------------------------------------------------------------
enum {
  IN_X = 0, IN_XALL, IN_MASK,
  IN_EMBX_W, IN_EMBX_B, IN_EMBA_W, IN_EMBA_B,
  IN_W1_W, IN_W1_B, IN_W2_W, IN_W2_B, IN_W3_W, IN_W3_B,
  IN_A1_W, IN_A1_B, IN_A2_W, IN_A2_B, IN_A3_W, IN_A3_B,
  IN_XI_W, IN_XI_B, IN_HI_W, IN_HI_B, IN_CI_W, IN_CI_B,
  IN_XF_W, IN_XF_B, IN_HF_W, IN_HF_B, IN_CF_W, IN_CF_B,
  IN_XC_W, IN_XC_B, IN_HC_W, IN_HC_B,
  IN_XO_W, IN_XO_B, IN_HO_W, IN_HO_B, IN_CO_W, IN_CO_B,
  IN_WQ_W, IN_WQ_B, IN_WREF_W, IN_WREF_B,
  IN_V, IN_WIH, IN_BIH, IN_WHH, IN_BHH,
  IN_H0, IN_C0, IN_R1, IN_R2, IN_R3
};

extern "C" void kernel_launch(void* const* d_in, const int* in_sizes, int n_in,
                              void* d_out, int out_size, void* d_ws,
                              size_t ws_size, hipStream_t stream) {
  (void)in_sizes; (void)n_in; (void)out_size; (void)ws_size;
  const int RS = kB * kS;  // 131072 flat context rows

  // ---- workspace carve-out -------------------------------------------------
  char* w = (char*)d_ws;
  auto alloc = [&](size_t bytes) -> void* {
    void* p = (void*)w;
    w += (bytes + 255) & ~(size_t)255;
    return p;
  };
  float* ctx   = (float*)alloc((size_t)RS * kDIM * 4);  // 134 MB
  bf16*  ctxbf = (bf16*)alloc((size_t)RS * kDIM * 2);   // 67 MB (LSTM operand)
  float* nz    = (float*)alloc((size_t)RS * 4);
  float* ntr   = (float*)alloc(kB * 4);
  float* xemb  = (float*)alloc(kB * kDIM * 4);
  float* hl    = (float*)alloc(kB * kDIM * 4);
  float* cl    = (float*)alloc(kB * kDIM * 4);
  float* qp    = (float*)alloc(kB * kDIM * 4);
  bf16* wih_t  = (bf16*)alloc((size_t)1024 * kDIM * 2);
  bf16* whh_t  = (bf16*)alloc((size_t)1024 * kDIM * 2);
  auto sq = [&]() { return (bf16*)alloc((size_t)kDIM * kDIM * 2); };
  bf16 *w1t = sq(), *w2t = sq(), *w3t = sq();
  bf16 *a1t = sq(), *a2t = sq(), *a3t = sq();
  bf16 *xit = sq(), *hit = sq(), *cit = sq();
  bf16 *xft = sq(), *hft = sq(), *cft = sq();
  bf16 *xct = sq(), *hct = sq();
  bf16 *xot = sq(), *hot = sq(), *cot = sq();
  bf16 *wqt = sq(), *wrt = sq();

  auto in = [&](int i) { return (const float*)d_in[i]; };

  // ---- weight prep (f32 [K][N] -> bf16 [N][K]) -----------------------------
  auto wt = [&](int idx, bf16* dst, int K, int N) {
    int total = K * N;
    k_wt<<<(total + 255) / 256, 256, 0, stream>>>(in(idx), dst, K, N);
  };
  wt(IN_WIH, wih_t, kDIM, 1024);
  wt(IN_WHH, whh_t, kDIM, 1024);
  wt(IN_W1_W, w1t, kDIM, kDIM);  wt(IN_W2_W, w2t, kDIM, kDIM);
  wt(IN_W3_W, w3t, kDIM, kDIM);
  wt(IN_A1_W, a1t, kDIM, kDIM);  wt(IN_A2_W, a2t, kDIM, kDIM);
  wt(IN_A3_W, a3t, kDIM, kDIM);
  wt(IN_XI_W, xit, kDIM, kDIM);  wt(IN_HI_W, hit, kDIM, kDIM);
  wt(IN_CI_W, cit, kDIM, kDIM);
  wt(IN_XF_W, xft, kDIM, kDIM);  wt(IN_HF_W, hft, kDIM, kDIM);
  wt(IN_CF_W, cft, kDIM, kDIM);
  wt(IN_XC_W, xct, kDIM, kDIM);  wt(IN_HC_W, hct, kDIM, kDIM);
  wt(IN_XO_W, xot, kDIM, kDIM);  wt(IN_HO_W, hot, kDIM, kDIM);
  wt(IN_CO_W, cot, kDIM, kDIM);
  wt(IN_WQ_W, wqt, kDIM, kDIM);  wt(IN_WREF_W, wrt, kDIM, kDIM);

  // ---- embeddings ----------------------------------------------------------
  k_embed<<<RS, 256, 0, stream>>>(in(IN_XALL), in(IN_EMBA_W), in(IN_EMBA_B),
                                  ctx, ctxbf, nz, 1);
  k_embed<<<kB, 256, 0, stream>>>(in(IN_X), in(IN_EMBX_W), in(IN_EMBX_B),
                                  xemb, nullptr, nullptr, 0);
  k_ntrue<<<kB, 256, 0, stream>>>(nz, ntr);

  // ---- LSTM scan -----------------------------------------------------------
  k_lstm<<<kB / 16, 256, 0, stream>>>(ctxbf, nz, wih_t, whh_t, in(IN_BIH),
                                      in(IN_BHH), in(IN_H0), in(IN_C0), hl, cl);

  // ---- 3 agg rounds (in place on ctx) --------------------------------------
  k_agg<<<RS / 128, 256, 0, stream>>>(ctx, w1t, in(IN_W1_B), a1t, in(IN_A1_B),
                                      in(IN_R1), nz, ntr);
  k_agg<<<RS / 128, 256, 0, stream>>>(ctx, w2t, in(IN_W2_B), a2t, in(IN_A2_B),
                                      in(IN_R2), nz, ntr);
  k_agg<<<RS / 128, 256, 0, stream>>>(ctx, w3t, in(IN_W3_B), a3t, in(IN_A3_B),
                                      in(IN_R3), nz, ntr);

  // ---- outputs layout: [softmax(B*S) | h(B*D) | c(B*D) | latent(B*S)] ------
  float* out     = (float*)d_out;
  float* out_sm  = out;
  float* out_h   = out + (size_t)kB * kS;
  float* out_c   = out_h + (size_t)kB * kDIM;
  float* out_lat = out_c + (size_t)kB * kDIM;

  // ---- encoder gates -------------------------------------------------------
  k_enc1<<<kB / 128, 256, 0, stream>>>(
      xemb, hl, cl, xit, in(IN_XI_B), hit, in(IN_HI_B), cit, in(IN_CI_B),
      xft, in(IN_XF_B), hft, in(IN_HF_B), cft, in(IN_CF_B),
      xct, in(IN_XC_B), hct, in(IN_HC_B), out_c);
  k_enc2<<<kB / 128, 256, 0, stream>>>(xemb, hl, out_c, xot, in(IN_XO_B), hot,
                                       in(IN_HO_B), cot, in(IN_CO_B), out_h);

  // ---- qp = h @ Wq + bq ----------------------------------------------------
  k_gemm<<<kB / 128, 256, 0, stream>>>(out_h, wqt, in(IN_WQ_B), qp, kDIM);

  // ---- fused refp + attention scores, then softmax -------------------------
  k_uref<<<RS / 128, 256, 0, stream>>>(ctx, wrt, in(IN_WREF_B), qp, in(IN_V),
                                       out_lat);
  k_softmax<<<kB, 256, 0, stream>>>(out_lat, in(IN_MASK), out_sm);
}
